// AMG_encoder_6141803233746
// MI455X (gfx1250) — compile-verified, matrix operand order checked by reading, not run on hardware
//
#include <hip/hip_runtime.h>
#include <stdint.h>

#define N_GRIDS  64
#define N_FEAT   2
#define GRES     64
#define NPTS     32      // points per block
#define NTHREADS 256     // 8 wave32s

typedef float        v2f  __attribute__((ext_vector_type(2)));
typedef float        v8f  __attribute__((ext_vector_type(8)));
typedef _Float16     v16h __attribute__((ext_vector_type(16)));
typedef unsigned int v4u  __attribute__((ext_vector_type(4)));
typedef int          v4i  __attribute__((ext_vector_type(4)));
typedef int          v8i  __attribute__((ext_vector_type(8)));

// ---- feature probes -------------------------------------------------------
#if __has_builtin(__builtin_amdgcn_tensor_load_to_lds) && __has_builtin(__builtin_amdgcn_s_wait_tensorcnt)
#define AMG_HAS_TDM 1
#if __has_include(<hip/amd_detail/amd_gfx1250_TDM.h>)
#define AMG_TDM_6ARG 1   // therock headers => 6-arg builtin
#endif
#endif

#if __has_builtin(__builtin_amdgcn_wmma_f32_16x16x4_f32)
#define AMG_WMMA_F32X4 1
#elif __has_builtin(__builtin_amdgcn_wmma_f32_16x16x32_f16)
#define AMG_WMMA_F16 1
#endif

__global__ __launch_bounds__(NTHREADS) void amg_encoder_kernel(
    const float* __restrict__ x,      // [B,3]
    const float* __restrict__ tm,     // [64,4,4]
    const float* __restrict__ grids,  // [64,2,64,64,64]
    float* __restrict__ out,          // [B,128]
    int batch)
{
  __shared__ float  s_tm[N_GRIDS * 16];     // 4 KB: transformation matrices
  __shared__ float4 s_pts[NPTS][N_GRIDS];   // 32 KB: transformed points (xyzw)

  const int tid  = threadIdx.x;
  const int lane = tid & 31;
  const int wave = tid >> 5;
  const int half = lane >> 4;   // 0: lanes 0-15, 1: lanes 16-31
  const int ml   = lane & 15;
  const int b0   = blockIdx.x * NPTS;

  // ---- Stage 0: stage 4KB of matrices into LDS ---------------------------
#if defined(AMG_HAS_TDM)
  if (wave == 0) {
    uint64_t ga      = (uint64_t)(uintptr_t)tm;
    uint32_t lds_off = (uint32_t)(uintptr_t)(&s_tm[0]);
    // D# group0: count=1, lds_addr, global_addr, type=2
    v4u g0 = { 1u,
               lds_off,
               (uint32_t)ga,
               (uint32_t)((ga >> 32) & 0x01FFFFFFu) | (2u << 30) };
    // D# group1: mask=0, data_size=2 (4B), 1-D tensor: dim0=tile0=stride0=1024
    v8i g1 = { (int)0x00020000,   // workgroup_mask=0 | data_size=2<<16
               (int)0x04000000,   // tensor_dim0[15:0]=1024 at bits[31:16]
               (int)0x00010000,   // tensor_dim0 hi=0 | tensor_dim1=1
               (int)0x04000000,   // tensor_dim1 hi=0 | tile_dim0=1024
               1,                 // tile_dim1=1 | tile_dim2=0
               1024,              // tensor_dim0_stride lo32
               0, 0 };
    v4i gz = { 0, 0, 0, 0 };
#if defined(AMG_TDM_6ARG)
    v8i gz8 = { 0, 0, 0, 0, 0, 0, 0, 0 };
    __builtin_amdgcn_tensor_load_to_lds(g0, g1, gz, gz, gz8, 0);
#else
    __builtin_amdgcn_tensor_load_to_lds(g0, g1, gz, gz, 0);
#endif
    __builtin_amdgcn_s_wait_tensorcnt(0);
  }
#else
  {
    const float4* src = (const float4*)tm;
    float4*       dst = (float4*)s_tm;
    dst[tid] = src[tid];          // 256 threads * 16B == 4KB
  }
#endif
  __syncthreads();

  // ---- Stage A: WMMA transforms: pts[g] = tm[g] * (x,y,z,1) --------------
  // Wave w owns grids [8w, 8w+8): 2 groups of 4 grids x 2 point tiles.
  // A (16x4 f32): lanes 0-15 hold K=0,1 (x,y); lanes 16-31 hold K=2,3 (z,1).
  // B (4x16 f32): column n -> grid gbase+(n>>2), row n&3; VGPR0/1 = K pair.
#pragma unroll
  for (int t = 0; t < 2; ++t) {
    int p  = t * 16 + ml;
    int gb = b0 + p; if (gb >= batch) gb = batch - 1;   // keep EXEC uniform
    const float* xp = x + 3 * (size_t)gb;
    float px = xp[0], py = xp[1], pz = xp[2];
#pragma unroll
    for (int G = 0; G < 2; ++G) {
      int gbase = wave * 8 + G * 4;
      int g = gbase + (ml >> 2);
      int r = ml & 3;
#if defined(AMG_WMMA_F32X4)
      v2f A;  A.x = half ? pz : px;  A.y = half ? 1.0f : py;
      const float* tp = s_tm + g * 16 + r * 4 + 2 * half;
      v2f Bv; Bv.x = tp[0]; Bv.y = tp[1];
      v8f C = {};
      v8f D = __builtin_amdgcn_wmma_f32_16x16x4_f32(
          false, A, false, Bv, (short)0, C, false, false);
#elif defined(AMG_WMMA_F16)
      // Fallback: embed the K=4 transform in K slots 0..3 of a 16x16x32 f16 WMMA.
      v16h A = {}; v16h Bv = {};
      if (half == 0) {
        A[0] = (_Float16)px; A[1] = (_Float16)py;
        A[2] = (_Float16)pz; A[3] = (_Float16)1.0f;
        const float* tq = s_tm + g * 16 + r * 4;
        Bv[0] = (_Float16)tq[0]; Bv[1] = (_Float16)tq[1];
        Bv[2] = (_Float16)tq[2]; Bv[3] = (_Float16)tq[3];
      }
      v8f C = {};
      v8f D = __builtin_amdgcn_wmma_f32_16x16x32_f16(
          false, A, false, Bv, (short)0, C, false, false);
#else
      // Pure-VALU fallback (no WMMA builtins available)
      float D[8];
      const float* tq = s_tm + g * 16 + r * 4;
#pragma unroll
      for (int v = 0; v < 8; ++v) {
        int pm  = t * 16 + v + 8 * half;
        int gb2 = b0 + pm; if (gb2 >= batch) gb2 = batch - 1;
        const float* xq = x + 3 * (size_t)gb2;
        D[v] = tq[0] * xq[0] + tq[1] * xq[1] + tq[2] * xq[2] + tq[3];
      }
#endif
      // D layout: VGPR v, lanes 0-15 -> M=v; lanes 16-31 -> M=v+8; N=ml.
      // Element (P, g, comp) sits at float offset (P*64+g)*4+comp; for this
      // tile that is base+ml (consecutive words -> conflict-free DS stores).
      float* dst = (float*)s_pts + (((t * 16 + 8 * half) * N_GRIDS + gbase) << 2) + ml;
#pragma unroll
      for (int v = 0; v < 8; ++v) dst[v * (N_GRIDS * 4)] = D[v];
    }
  }
  __syncthreads();

  // ---- Stage B: trilinear gather, L2-resident grids, NT output -----------
  const size_t CH = (size_t)GRES * GRES * GRES;   // 262144 elems per channel
#pragma unroll
  for (int it = 0; it < 8; ++it) {
    int p = wave * 4 + (it >> 1);
    int g = lane + 32 * (it & 1);
    float4 pt = s_pts[p][g];                      // stride-16B, coalesced DS read
    int bb = b0 + p;

    float ix = (pt.x + 1.0f) * 31.5f;
    float iy = (pt.y + 1.0f) * 31.5f;
    float iz = (pt.z + 1.0f) * 31.5f;
    float xf = floorf(ix), yf = floorf(iy), zf = floorf(iz);
    float fx = ix - xf, fy = iy - yf, fz = iz - zf;
    int x0 = (int)xf, y0 = (int)yf, z0 = (int)zf;
    float wx[2] = { 1.0f - fx, fx };
    float wy[2] = { 1.0f - fy, fy };
    float wz[2] = { 1.0f - fz, fz };

    const float* gp = grids + (size_t)g * (N_FEAT * CH);
    float acc0 = 0.0f, acc1 = 0.0f;
#pragma unroll
    for (int dz = 0; dz < 2; ++dz) {
      int zi = z0 + dz; int zc = min(max(zi, 0), GRES - 1);
      bool zv = ((unsigned)zi < (unsigned)GRES);
#pragma unroll
      for (int dy = 0; dy < 2; ++dy) {
        int yi = y0 + dy; int yc = min(max(yi, 0), GRES - 1);
        bool yv = ((unsigned)yi < (unsigned)GRES);
#pragma unroll
        for (int dx = 0; dx < 2; ++dx) {
          int xi = x0 + dx; int xc = min(max(xi, 0), GRES - 1);
          bool xv = ((unsigned)xi < (unsigned)GRES);
          float w = wz[dz] * wy[dy] * wx[dx];
          w = (zv && yv && xv) ? w : 0.0f;       // zeros padding, branch-free
          int idx = (zc << 12) | (yc << 6) | xc;
          acc0 = fmaf(gp[idx], w, acc0);
          acc1 = fmaf(gp[CH + idx], w, acc1);
        }
      }
    }
    if (bb < batch) {
      v2f o; o.x = acc0; o.y = acc1;
      // Non-temporal: stream the 134MB output past L2 so the feature grids
      // (134MB) stay resident in the 192MB L2 for the random gathers.
      __builtin_nontemporal_store(o,
          (v2f*)(out + (size_t)bb * (N_GRIDS * N_FEAT)) + g);
    }
  }
}

extern "C" void kernel_launch(void* const* d_in, const int* in_sizes, int n_in,
                              void* d_out, int out_size, void* d_ws, size_t ws_size,
                              hipStream_t stream) {
  const float* x     = (const float*)d_in[0];
  const float* tm    = (const float*)d_in[1];
  const float* grids = (const float*)d_in[2];
  float*       out   = (float*)d_out;
  int batch  = in_sizes[0] / 3;
  int blocks = (batch + NPTS - 1) / NPTS;
  hipLaunchKernelGGL(amg_encoder_kernel, dim3(blocks), dim3(NTHREADS), 0, stream,
                     x, tm, grids, out, batch);
}
